// GPT_29386166239738
// MI455X (gfx1250) — matrix operand order repaired and checked by reading, not compiled, not run on hardware
//
#include <hip/hip_runtime.h>

#define Bb 4
#define Tt 1024
#define Ee 1024
#define Hh 16
#define HS 64
#define Ll 12
#define Vv 32000
#define FF 4096
#define Mm (Bb*Tt)

typedef __attribute__((ext_vector_type(16))) __bf16 v16bf;
typedef __attribute__((ext_vector_type(8)))  float  v8f;

union Frag { unsigned u[8]; __bf16 h[16]; v16bf v; };

__device__ __forceinline__ v8f zero8() { v8f z = {0.f,0.f,0.f,0.f,0.f,0.f,0.f,0.f}; return z; }

__device__ __forceinline__ unsigned lds_off(const void* p) {
    // LDS aperture: low 32 bits of a flat LDS address are the LDS byte offset.
    return (unsigned)(size_t)p;
}
__device__ __forceinline__ void async_b32(unsigned lds, const float* g) {
    unsigned long long ga = (unsigned long long)(size_t)g;
    asm volatile("global_load_async_to_lds_b32 %0, %1, off" :: "v"(lds), "v"(ga) : "memory");
}
__device__ __forceinline__ void async_b128(unsigned lds, const float* g) {
    unsigned long long ga = (unsigned long long)(size_t)g;
    asm volatile("global_load_async_to_lds_b128 %0, %1, off" :: "v"(lds), "v"(ga) : "memory");
}

// ---------------------------------------------------------------------------
// Embedding: x[b,t,:] = tok_emb[idx[b,t],:] + pos_emb[t,:]
// ---------------------------------------------------------------------------
__global__ void embed_kernel(const int* __restrict__ idx,
                             const float* __restrict__ tok,
                             const float* __restrict__ pos,
                             float* __restrict__ x) {
    size_t i = (size_t)blockIdx.x * blockDim.x + threadIdx.x;
    if (i >= (size_t)Mm * Ee) return;
    int e  = (int)(i & (Ee - 1));
    int bt = (int)(i >> 10);        // Ee == 1024
    int t  = bt & (Tt - 1);
    int token = idx[bt];
    x[i] = tok[(size_t)token * Ee + e] + pos[(size_t)t * Ee + e];
}

// ---------------------------------------------------------------------------
// LayerNorm over E=1024, one block (256 threads) per row
// ---------------------------------------------------------------------------
__global__ __launch_bounds__(256)
void ln_kernel(const float* __restrict__ x, const float* __restrict__ g,
               const float* __restrict__ bta, float* __restrict__ y) {
    __shared__ float red[256];
    const int row = blockIdx.x;
    const int tid = threadIdx.x;
    const float* xr = x + (size_t)row * Ee;
    float lv[4];
    float s = 0.f;
    #pragma unroll
    for (int i = 0; i < 4; i++) { lv[i] = xr[tid + 256 * i]; s += lv[i]; }
    red[tid] = s; __syncthreads();
    for (int off = 128; off; off >>= 1) { if (tid < off) red[tid] += red[tid + off]; __syncthreads(); }
    float mean = red[0] * (1.0f / Ee);
    __syncthreads();
    float vs = 0.f;
    #pragma unroll
    for (int i = 0; i < 4; i++) { float d = lv[i] - mean; vs += d * d; }
    red[tid] = vs; __syncthreads();
    for (int off = 128; off; off >>= 1) { if (tid < off) red[tid] += red[tid + off]; __syncthreads(); }
    float rstd = rsqrtf(red[0] * (1.0f / Ee) + 1e-5f);
    float* yr = y + (size_t)row * Ee;
    #pragma unroll
    for (int i = 0; i < 4; i++) {
        int c = tid + 256 * i;
        yr[c] = (lv[i] - mean) * rstd * g[c] + bta[c];
    }
}

// ---------------------------------------------------------------------------
// bf16-WMMA GEMM:  C(m,n) = sum_k A(m,k)*B(k,n) [+bias][+Res][relu]
// Block tile 128(M) x 128(N), 8 waves (4x2), wave tile 32x64, K step 32.
// Software-pipelined: double-buffered LDS, GLOBAL_LOAD_ASYNC_TO_LDS DMA for
// tile t+1 runs under the WMMAs of tile t (s_wait_asynccnt 0x14 keeps the
// in-flight tile's 20 transfers outstanding).  B transposed in flight via
// per-lane LDS scatter.  bf16 conversion during fragment build.
//
// Normal mode (hmode==0): B(k,n) = Bw[z*b_z + k*sbk + n*sbn]
// Head mode  (hmode!=0): column n -> head n>>6, channel n&63:
//                        B(k,n) = Bw[z*b_z + (n>>6)*hz + k*sbk + (n&63)*sbn]
//                        (two heads packed per 128-wide tile; no masking)
// C fp32: C[z*c_z + m*ldc + n].  M % 128 == 0, K % 32 == 0, N % 128 == 0.
// ---------------------------------------------------------------------------
__global__ __launch_bounds__(256)
void gemm_bf16(const float* __restrict__ A, int lda,
               const float* __restrict__ Bw, long b_z, int sbk, int sbn,
               int hmode, long hz,
               const float* __restrict__ bias,
               const float* __restrict__ Res,
               float* __restrict__ C, long c_z, int ldc,
               int K, int relu)
{
    __shared__ float Af[2 * 128 * 36];   // [buf][m][k], pad 36 -> 144B rows (16B aligned)
    __shared__ float Bf[2 * 128 * 36];   // [buf][n][k] (transposed)
    const int tid  = threadIdx.x;
    const int lane = tid & 31;
    const int wv   = tid >> 5;
    const int wr   = wv >> 1;          // wave row 0..3 (32 rows each)
    const int wc   = wv & 1;           // wave col 0..1 (64 cols each)
    const int lm   = lane & 15;
    const int lh   = lane >> 4;
    const int mb   = blockIdx.y * 128;
    const int nb   = blockIdx.x * 128;
    const int z    = blockIdx.z;
    const float* Bz = Bw + (long)z * b_z;
    float* Cz = C + (long)z * c_z;
    const float* Rz = Res ? (Res + (long)z * c_z) : (const float*)0;

    v8f acc[2][4];
    #pragma unroll
    for (int i = 0; i < 2; i++)
        #pragma unroll
        for (int s = 0; s < 4; s++) acc[i][s] = zero8();

    // async copy mappings
    const int arow = tid >> 1;             // A: 128 rows x 32 k; 16 floats/thread (4 x b128)
    const int akk  = (tid & 1) * 16;
    const int bkk  = tid >> 3;             // B: 32 k x 128 n; 16 scalar scatters/thread
    const int bn0  = (tid & 7) * 16;       // 16-col chunk never crosses a 64 boundary

    // per-thread B column-base offset (head-grouped or plain)
    size_t coloff;
    if (hmode) coloff = (size_t)((nb + bn0) >> 6) * (size_t)hz
                      + (size_t)((nb + bn0) & 63) * (size_t)sbn;
    else       coloff = (size_t)(nb + bn0) * (size_t)sbn;

    // issue one K-tile's async DMA (20 transfers per thread)
    auto issue_tile = [&](int k0, int buf) {
        float* Afb = &Af[buf * (128 * 36)];
        float* Bfb = &Bf[buf * (128 * 36)];
        const float* ap = A + (size_t)(mb + arow) * lda + (k0 + akk);
        if (k0 + 64 < K) __builtin_prefetch(ap + 64, 0, 0);
        unsigned l0 = lds_off(&Afb[arow * 36 + akk]);
        #pragma unroll
        for (int i = 0; i < 4; i++) async_b128(l0 + i * 16, ap + i * 4);
        const float* bp = Bz + (size_t)(k0 + bkk) * sbk + coloff;
        #pragma unroll
        for (int i = 0; i < 16; i++)
            async_b32(lds_off(&Bfb[(bn0 + i) * 36 + bkk]), bp + (size_t)i * sbn);
    };

    const int NT = K >> 5;
    issue_tile(0, 0);
    for (int t = 0; t < NT; t++) {
        const int cur = t & 1;
        if (t + 1 < NT) {
            issue_tile((t + 1) << 5, (t + 1) & 1);
            asm volatile("s_wait_asynccnt 0x14" ::: "memory");  // tile t done, t+1 in flight
        } else {
            asm volatile("s_wait_asynccnt 0x0" ::: "memory");
        }
        __syncthreads();

        const float* Afb = &Af[cur * (128 * 36)];
        const float* Bfb = &Bf[cur * (128 * 36)];

        // ---- fragments (fp32 LDS -> bf16 regs) ----
        Frag af[2];
        #pragma unroll
        for (int i = 0; i < 2; i++) {
            const float* ar = &Afb[(wr * 32 + i * 16 + lm) * 36];
            #pragma unroll
            for (int j = 0; j < 4; j++) {
                float2 p = *(const float2*)(ar + 8 * lh + 2 * j);
                af[i].h[2 * j]     = (__bf16)p.x;
                af[i].h[2 * j + 1] = (__bf16)p.y;
            }
            #pragma unroll
            for (int j = 0; j < 4; j++) {
                float2 p = *(const float2*)(ar + 16 + 8 * lh + 2 * j);
                af[i].h[8 + 2 * j]     = (__bf16)p.x;
                af[i].h[8 + 2 * j + 1] = (__bf16)p.y;
            }
        }
        Frag bf[4];
        #pragma unroll
        for (int s = 0; s < 4; s++) {
            const float* br = &Bfb[(wc * 64 + s * 16 + lm) * 36 + lh * 8];
            #pragma unroll
            for (int j = 0; j < 8; j++) {
                float2 p = *(const float2*)(br + 2 * j + (j >> 2) * 8);
                bf[s].h[2 * j]     = (__bf16)p.x;
                bf[s].h[2 * j + 1] = (__bf16)p.y;
            }
        }
        // ---- 8 WMMAs per K-step ----
        #pragma unroll
        for (int i = 0; i < 2; i++)
            #pragma unroll
            for (int s = 0; s < 4; s++)
                acc[i][s] = __builtin_amdgcn_wmma_f32_16x16x32_bf16(
                                false, af[i].v, false, bf[s].v, (short)0, acc[i][s], false, false);
        __syncthreads();
    }

    // ---- epilogue: bias + residual + relu ----
    #pragma unroll
    for (int s = 0; s < 4; s++) {
        int ncol = nb + wc * 64 + s * 16 + lm;
        float bv = bias ? bias[ncol] : 0.f;
        #pragma unroll
        for (int i = 0; i < 2; i++) {
            #pragma unroll
            for (int r = 0; r < 8; r++) {
                int mrow  = mb + wr * 32 + i * 16 + r + 8 * lh;
                float val = acc[i][s][r] + bv;
                if (Rz) val += Rz[(size_t)mrow * ldc + ncol];
                if (relu) val = fmaxf(val, 0.f);
                Cz[(size_t)mrow * ldc + ncol] = val;
            }
        }
    }
}

// ---------------------------------------------------------------------------
// Fused causal attention (flash-style, online softmax), one wave32 per
// (b, h, 16-row q tile).  scores = (q @ k^T) * sqrt(HS)  [reference bug kept].
// q,k,v,o layout: [B, T, H, HS] fp32.  V tile DMA'd via async-to-LDS scatter,
// overlapped with score WMMAs + online softmax.
// ---------------------------------------------------------------------------
__global__ __launch_bounds__(32)
void attn_fused(const float* __restrict__ q, const float* __restrict__ k,
                const float* __restrict__ v, float* __restrict__ o)
{
    __shared__ __bf16 Pl[16 * 34];
    __shared__ float  Vf[64 * 34];     // [d][s_local] fp32 (transposed in flight)
    const int lane = threadIdx.x;
    const int lm = lane & 15, lh = lane >> 4;
    const int mb = blockIdx.x * 16;
    const int h  = blockIdx.y;
    const int b  = blockIdx.z;

    // Q A-fragments for K=64 (two K-steps of 32)
    Frag qf[2];
    {
        const float* qr = q + ((size_t)(b * Tt + mb + lm) * Hh + h) * HS;
        #pragma unroll
        for (int ks = 0; ks < 2; ks++) {
            #pragma unroll
            for (int j = 0; j < 4; j++) {
                int d = ks * 32 + 8 * lh + 2 * j;
                qf[ks].h[2 * j]     = (__bf16)qr[d];
                qf[ks].h[2 * j + 1] = (__bf16)qr[d + 1];
            }
            #pragma unroll
            for (int j = 0; j < 4; j++) {
                int d = ks * 32 + 16 + 8 * lh + 2 * j;
                qf[ks].h[8 + 2 * j]     = (__bf16)qr[d];
                qf[ks].h[8 + 2 * j + 1] = (__bf16)qr[d + 1];
            }
        }
    }

    float mstate[8], lstate[8];
    v8f   oacc[4];
    #pragma unroll
    for (int r = 0; r < 8; r++) { mstate[r] = -3.0e38f; lstate[r] = 0.f; }
    #pragma unroll
    for (int s = 0; s < 4; s++) oacc[s] = zero8();

    const float scale = 8.0f;  // sqrt(HS); reference MULTIPLIES

    for (int sb = 0; sb <= mb; sb += 32) {
        // ---- kick off V tile DMA: Vf[d][s_local] (transpose scatter) ----
        {
            int srow = sb + lane;
            srow = (srow < Tt) ? srow : (Tt - 1);
            const float* vr = v + ((size_t)(b * Tt + srow) * Hh + h) * HS;
            #pragma unroll
            for (int d = 0; d < 64; d++)
                async_b32(lds_off(&Vf[d * 34 + lane]), vr + d);
        }
        // ---- scores for 32 columns: two 16x16 WMMA C tiles ----
        v8f sc[2];
        sc[0] = zero8(); sc[1] = zero8();
        #pragma unroll
        for (int half = 0; half < 2; half++) {
            int srow = sb + half * 16 + lm;
            srow = (srow < Tt) ? srow : (Tt - 1);
            const float* kr = k + ((size_t)(b * Tt + srow) * Hh + h) * HS;
            const int ko = lh * 8;
            #pragma unroll
            for (int ks = 0; ks < 2; ks++) {
                Frag bfrag;
                #pragma unroll
                for (int j = 0; j < 8; j++) {
                    int d = ks * 32 + ko + 2 * j + (j >> 2) * 8;
                    bfrag.h[2 * j]     = (__bf16)kr[d];
                    bfrag.h[2 * j + 1] = (__bf16)kr[d + 1];
                }
                sc[half] = __builtin_amdgcn_wmma_f32_16x16x32_bf16(
                               false, qf[ks].v, false, bfrag.v, (short)0, sc[half], false, false);
            }
        }
        // ---- scale + causal mask ----
        const int rowbase = mb + 8 * lh;
        #pragma unroll
        for (int half = 0; half < 2; half++) {
            int col = sb + half * 16 + lm;
            #pragma unroll
            for (int r = 0; r < 8; r++) {
                float xv = sc[half][r] * scale;
                sc[half][r] = (col <= rowbase + r) ? xv : -1.0e30f;
            }
        }
        // ---- online softmax: row max, rescale, exp, row sum ----
        float mnew[8], alpha[8];
        #pragma unroll
        for (int r = 0; r < 8; r++) {
            float mx = fmaxf(sc[0][r], sc[1][r]);
            #pragma unroll
            for (int off = 1; off < 16; off <<= 1) mx = fmaxf(mx, __shfl_xor(mx, off, 32));
            mnew[r]  = fmaxf(mstate[r], mx);
            alpha[r] = __expf(mstate[r] - mnew[r]);
            mstate[r] = mnew[r];
        }
        #pragma unroll
        for (int r = 0; r < 8; r++) {
            float p0 = __expf(sc[0][r] - mnew[r]);
            float p1 = __expf(sc[1][r] - mnew[r]);
            float sm = p0 + p1;
            #pragma unroll
            for (int off = 1; off < 16; off <<= 1) sm += __shfl_xor(sm, off, 32);
            lstate[r] = lstate[r] * alpha[r] + sm;
            Pl[(r + 8 * lh) * 34 + lm]      = (__bf16)p0;
            Pl[(r + 8 * lh) * 34 + 16 + lm] = (__bf16)p1;
            #pragma unroll
            for (int s = 0; s < 4; s++) oacc[s][r] *= alpha[r];
        }
        asm volatile("s_wait_asynccnt 0x0" ::: "memory");
        __syncthreads();
        // ---- P A-fragment from LDS ----
        Frag pf;
        {
            const __bf16* pr = &Pl[lm * 34];
            #pragma unroll
            for (int j = 0; j < 4; j++) pf.u[j]     = *(const unsigned*)&pr[8 * lh + 2 * j];
            #pragma unroll
            for (int j = 0; j < 4; j++) pf.u[4 + j] = *(const unsigned*)&pr[16 + 8 * lh + 2 * j];
        }
        // ---- O += P @ V : 4 d-subtiles ----
        #pragma unroll
        for (int s = 0; s < 4; s++) {
            Frag vfrag;
            const float* vrr = &Vf[(s * 16 + lm) * 34 + lh * 8];
            #pragma unroll
            for (int j = 0; j < 8; j++) {
                float2 p = *(const float2*)(vrr + 2 * j + (j >> 2) * 8);
                vfrag.h[2 * j]     = (__bf16)p.x;
                vfrag.h[2 * j + 1] = (__bf16)p.y;
            }
            oacc[s] = __builtin_amdgcn_wmma_f32_16x16x32_bf16(
                          false, pf.v, false, vfrag.v, (short)0, oacc[s], false, false);
        }
        __syncthreads();
    }

    // ---- normalize and store O ----
    #pragma unroll
    for (int r = 0; r < 8; r++) {
        float inv = 1.0f / lstate[r];
        int row = mb + r + 8 * lh;
        float* orow = o + ((size_t)(b * Tt + row) * Hh + h) * HS;
        #pragma unroll
        for (int s = 0; s < 4; s++) orow[s * 16 + lm] = oacc[s][r] * inv;
    }
}

// ---------------------------------------------------------------------------
// Loss: mean NLL of log_softmax(logits) at targets
// ---------------------------------------------------------------------------
__global__ void zero_kernel(float* p) { *p = 0.f; }

__global__ __launch_bounds__(256)
void loss_kernel(const float* __restrict__ logits, const int* __restrict__ targets,
                 float* __restrict__ loss) {
    __shared__ float red[256];
    const int row = blockIdx.x;
    const int tid = threadIdx.x;
    const float* lr = logits + (size_t)row * Vv;
    float mx = -3.0e38f;
    for (int i = tid; i < Vv; i += 256) mx = fmaxf(mx, lr[i]);
    red[tid] = mx; __syncthreads();
    for (int off = 128; off; off >>= 1) { if (tid < off) red[tid] = fmaxf(red[tid], red[tid + off]); __syncthreads(); }
    mx = red[0]; __syncthreads();
    float sm = 0.f;
    for (int i = tid; i < Vv; i += 256) sm += __expf(lr[i] - mx);
    red[tid] = sm; __syncthreads();
    for (int off = 128; off; off >>= 1) { if (tid < off) red[tid] += red[tid + off]; __syncthreads(); }
    if (tid == 0) {
        float lse = mx + __logf(red[0]);
        float nll = lse - lr[targets[row]];
        atomicAdd(loss, nll * (1.0f / (float)Mm));
    }
}

// ---------------------------------------------------------------------------
// Host-side orchestration
// ---------------------------------------------------------------------------
extern "C" void kernel_launch(void* const* d_in, const int* in_sizes, int n_in,
                              void* d_out, int out_size, void* d_ws, size_t ws_size,
                              hipStream_t stream)
{
    (void)in_sizes; (void)n_in; (void)out_size; (void)ws_size;
    const int*   idx     = (const int*)d_in[0];
    const int*   targets = (const int*)d_in[1];
    const float* tok_emb = (const float*)d_in[2];
    const float* pos_emb = (const float*)d_in[3];
    const float* Wq      = (const float*)d_in[4];
    const float* Wk      = (const float*)d_in[5];
    const float* Wv      = (const float*)d_in[6];
    const float* Wp      = (const float*)d_in[7];
    const float* bp      = (const float*)d_in[8];
    const float* ln1_g   = (const float*)d_in[9];
    const float* ln1_b   = (const float*)d_in[10];
    const float* ln2_g   = (const float*)d_in[11];
    const float* ln2_b   = (const float*)d_in[12];
    const float* W1      = (const float*)d_in[13];
    const float* b1      = (const float*)d_in[14];
    const float* W2      = (const float*)d_in[15];
    const float* b2      = (const float*)d_in[16];
    const float* lnf_g   = (const float*)d_in[17];
    const float* lnf_b   = (const float*)d_in[18];
    const float* Wlm     = (const float*)d_in[19];
    const float* blm     = (const float*)d_in[20];

    float* logits = (float*)d_out;
    float* loss   = logits + (size_t)Mm * Vv;

    float* ws = (float*)d_ws;
    const size_t ME = (size_t)Mm * Ee;
    float* x  = ws;
    float* h  = x  + ME;
    float* qb = h  + ME;
    float* kb = qb + ME;
    float* vb = kb + ME;
    float* ob = vb + ME;
    float* yb = ob + ME;   // [M, FF]

    embed_kernel<<<(Mm * Ee + 255) / 256, 256, 0, stream>>>(idx, tok_emb, pos_emb, x);

    for (int l = 0; l < Ll; l++) {
        const float* Wq_l = Wq + (size_t)l * Hh * Ee * HS;
        const float* Wk_l = Wk + (size_t)l * Hh * Ee * HS;
        const float* Wv_l = Wv + (size_t)l * Hh * Ee * HS;
        const float* Wp_l = Wp + (size_t)l * Ee * Ee;
        const float* W1_l = W1 + (size_t)l * Ee * FF;
        const float* W2_l = W2 + (size_t)l * FF * Ee;

        ln_kernel<<<Mm, 256, 0, stream>>>(x, ln1_g + (size_t)l * Ee, ln1_b + (size_t)l * Ee, h);

        // q/k/v: head-grouped GEMMs, 2 heads per 128-wide tile, grid.z = H/2.
        // B(k, n) = W[head][e][d]: z stride = 2*E*HS, head stride hz = E*HS,
        // sbk = HS, sbn = 1.  C: out col = z*128 + n  (c_z = 128, ldc = E).
        dim3 gqkv(1, Mm / 128, Hh / 2);
        gemm_bf16<<<gqkv, 256, 0, stream>>>(h, Ee, Wq_l, (long)2 * Ee * HS, HS, 1,
                                            1, (long)Ee * HS,
                                            nullptr, nullptr, qb, (long)128, Ee, Ee, 0);
        gemm_bf16<<<gqkv, 256, 0, stream>>>(h, Ee, Wk_l, (long)2 * Ee * HS, HS, 1,
                                            1, (long)Ee * HS,
                                            nullptr, nullptr, kb, (long)128, Ee, Ee, 0);
        gemm_bf16<<<gqkv, 256, 0, stream>>>(h, Ee, Wv_l, (long)2 * Ee * HS, HS, 1,
                                            1, (long)Ee * HS,
                                            nullptr, nullptr, vb, (long)128, Ee, Ee, 0);

        attn_fused<<<dim3(Tt / 16, Hh, Bb), 32, 0, stream>>>(qb, kb, vb, ob);

        // x = x + o @ Wp + bp
        gemm_bf16<<<dim3(Ee / 128, Mm / 128, 1), 256, 0, stream>>>(
            ob, Ee, Wp_l, 0, Ee, 1, 0, 0,
            bp + (size_t)l * Ee, x, x, 0, Ee, Ee, 0);

        ln_kernel<<<Mm, 256, 0, stream>>>(x, ln2_g + (size_t)l * Ee, ln2_b + (size_t)l * Ee, h);

        // y = relu(h @ W1 + b1)
        gemm_bf16<<<dim3(FF / 128, Mm / 128, 1), 256, 0, stream>>>(
            h, Ee, W1_l, 0, FF, 1, 0, 0,
            b1 + (size_t)l * FF, nullptr, yb, 0, FF, Ee, 1);

        // x = x + y @ W2 + b2
        gemm_bf16<<<dim3(Ee / 128, Mm / 128, 1), 256, 0, stream>>>(
            yb, FF, W2_l, 0, Ee, 1, 0, 0,
            b2 + (size_t)l * Ee, x, x, 0, Ee, FF, 0);
    }

    ln_kernel<<<Mm, 256, 0, stream>>>(x, lnf_g, lnf_b, h);

    // logits = h @ Wlm + blm
    gemm_bf16<<<dim3(Vv / 128, Mm / 128, 1), 256, 0, stream>>>(
        h, Ee, Wlm, 0, Vv, 1, 0, 0,
        blm, nullptr, logits, 0, Vv, Ee, 0);

    zero_kernel<<<1, 1, 0, stream>>>(loss);
    loss_kernel<<<Mm, 256, 0, stream>>>(logits, targets, loss);
}
